// TriangleMultiplicationOutgoing_91207925498266
// MI455X (gfx1250) — matrix operand order repaired and checked by reading, not compiled, not run on hardware
//
#include <hip/hip_runtime.h>
#include <hip/hip_bf16.h>
#include <stdint.h>

typedef __bf16 bf16_t;
typedef __attribute__((ext_vector_type(16))) __bf16 v16bf;
typedef __attribute__((ext_vector_type(8)))  __bf16 v8bf;
typedef __attribute__((ext_vector_type(8)))  float  v8f;

#define NN   512
#define CC   128
#define NROW (NN * NN)          // 262144 flattened (i,j) rows
#define LN_EPS 1e-5f

__device__ __forceinline__ float sigmoidf_(float x) { return 1.0f / (1.0f + __expf(-x)); }

// ---- WMMA fragment loaders (wave32, 16x16x32 bf16) ------------------------
// A-matrix 16x32: lane -> M = lane%16; half = lane/16;
//   element e holds K = 16*half + e (contiguous 16)
__device__ __forceinline__ v16bf fragA_contig(const bf16_t* p, int half) {
  v8bf lo = *(const v8bf*)(p + 16 * half);
  v8bf hi = *(const v8bf*)(p + 16 * half + 8);
  return __builtin_shufflevector(lo, hi, 0,1,2,3,4,5,6,7,8,9,10,11,12,13,14,15);
}
// B-matrix 32x16: lane -> N = lane%16; half = lane/16; element e holds K = 16*half + e
__device__ __forceinline__ v16bf fragB_contig(const bf16_t* p, int half) {
  v8bf lo = *(const v8bf*)(p + 16 * half);
  v8bf hi = *(const v8bf*)(p + 16 * half + 8);
  return __builtin_shufflevector(lo, hi, 0,1,2,3,4,5,6,7,8,9,10,11,12,13,14,15);
}
// Pre-swizzled weight fragment: tile (kb,nb) -> 32 lanes x 16 contiguous halfs
__device__ __forceinline__ v16bf fragW(const bf16_t* wp, int kb, int nb, int lane) {
  const bf16_t* p = wp + ((((nb << 2) + kb) << 5) + lane) * 16;
  v8bf lo = *(const v8bf*)p;
  v8bf hi = *(const v8bf*)(p + 8);
  return __builtin_shufflevector(lo, hi, 0,1,2,3,4,5,6,7,8,9,10,11,12,13,14,15);
}

// ---- CDNA5 async global->LDS copy (per-lane 16B), ASYNCcnt-tracked --------
__device__ __forceinline__ void async_copy_b128(uint32_t lds_off, const bf16_t* gptr) {
  asm volatile("global_load_async_to_lds_b128 %0, %1, off"
               :: "v"(lds_off), "v"((uint64_t)(uintptr_t)gptr) : "memory");
}

// ---- Kernel 0: convert + swizzle the four 128x128 weights into -----------
//      WMMA-B fragment-major bf16 layout: idx = ((nb*4+kb)*32 + lane)*16 + e
//      holding W[kb*32 + 16*(lane/16) + e][nb*16 + lane%16]
__global__ void k_pack_weights(const float* __restrict__ Wa, const float* __restrict__ Wb,
                               const float* __restrict__ Wg, const float* __restrict__ Wo,
                               bf16_t* __restrict__ wab, bf16_t* __restrict__ wbb,
                               bf16_t* __restrict__ wgb, bf16_t* __restrict__ wob) {
  int id = blockIdx.x * blockDim.x + threadIdx.x;   // 0..65535
  int m  = id >> 14;
  int i  = id & 16383;
  int e  = i & 15, l = (i >> 4) & 31, tl = i >> 9;
  int kb = tl & 3, nb = tl >> 2;
  int K   = kb * 32 + ((l >> 4) << 4) + e;          // B-operand K pattern
  int col = (nb << 4) + (l & 15);
  const float* src = (m == 0) ? Wa : (m == 1) ? Wb : (m == 2) ? Wg : Wo;
  bf16_t*      dst = (m == 0) ? wab : (m == 1) ? wbb : (m == 2) ? wgb : wob;
  dst[i] = (bf16_t)src[K * CC + col];
}

// ---- Kernel 1: fused LayerNorm + {Wa,Wb,Wg} projections + SiLU/sigmoid ----
__global__ void k_ln_proj(const float* __restrict__ pair,
                          const float* __restrict__ nw, const float* __restrict__ nb,
                          const bf16_t* __restrict__ WaB, const bf16_t* __restrict__ WbB,
                          const bf16_t* __restrict__ WgB,
                          bf16_t* __restrict__ aws, bf16_t* __restrict__ bws,
                          float* __restrict__ gws) {
  __shared__ float  psum[16][16];
  __shared__ float  psq[16][16];
  __shared__ float  mu_s[16], rs_s[16];
  __shared__ bf16_t zb[16][CC];          // normalized rows, bf16
  __shared__ float  trb[2][8][16][16];   // transpose staging for a,b tiles

  const int  t       = threadIdx.x;
  const int  r       = t >> 4, q = t & 15;
  const long rowBase = (long)blockIdx.x * 16;

  const float* src = pair + (rowBase + r) * CC + q * 8;
  float x[8], s = 0.f, sq = 0.f;
#pragma unroll
  for (int u = 0; u < 8; u++) { x[u] = src[u]; s += x[u]; sq += x[u] * x[u]; }
  psum[r][q] = s; psq[r][q] = sq;
  __syncthreads();
  if (t < 16) {
    float ss = 0.f, qq = 0.f;
    for (int u = 0; u < 16; u++) { ss += psum[t][u]; qq += psq[t][u]; }
    float mu  = ss * (1.0f / CC);
    float var = qq * (1.0f / CC) - mu * mu;
    mu_s[t] = mu;
    rs_s[t] = rsqrtf(var + LN_EPS);
  }
  __syncthreads();
  {
    float mu = mu_s[r], rs = rs_s[r];
#pragma unroll
    for (int u = 0; u < 8; u++) {
      int c = q * 8 + u;
      zb[r][c] = (bf16_t)((x[u] - mu) * rs * nw[c] + nb[c]);
    }
  }
  __syncthreads();

  const int w = t >> 5, lane = t & 31, half = lane >> 4, nn = lane & 15;
  v8f accA = {}, accB = {}, accG = {};
#pragma unroll
  for (int kb = 0; kb < 4; kb++) {
    v16bf af  = fragA_contig(&zb[nn][kb * 32], half);
    v16bf bfa = fragW(WaB, kb, w, lane);
    v16bf bfb = fragW(WbB, kb, w, lane);
    v16bf bfg = fragW(WgB, kb, w, lane);
    accA = __builtin_amdgcn_wmma_f32_16x16x32_bf16(false, af, false, bfa, (short)0, accA, false, false);
    accB = __builtin_amdgcn_wmma_f32_16x16x32_bf16(false, af, false, bfb, (short)0, accB, false, false);
    accG = __builtin_amdgcn_wmma_f32_16x16x32_bf16(false, af, false, bfg, (short)0, accG, false, false);
  }

  v8f va, vb;
#pragma unroll
  for (int rr = 0; rr < 8; rr++) {
    float xa = accA[rr]; va[rr] = xa * sigmoidf_(xa);
    float xb = accB[rr]; vb[rr] = xb * sigmoidf_(xb);
    long row = rowBase + rr + 8 * half;
    gws[row * CC + w * 16 + nn] = sigmoidf_(accG[rr]);
  }
#pragma unroll
  for (int rr = 0; rr < 8; rr++) {
    trb[0][w][rr + 8 * half][nn] = va[rr];
    trb[1][w][rr + 8 * half][nn] = vb[rr];
  }
  __syncthreads();
  {
    int tensor = t >> 7;
    int cidx   = t & 127;
    int ww     = cidx >> 4, nc = cidx & 15;
    bf16_t* dst = (tensor ? bws : aws) + (long)cidx * NROW + rowBase;
    v8bf v0, v1;
#pragma unroll
    for (int m = 0; m < 8; m++) v0[m] = (bf16_t)trb[tensor][ww][m][nc];
#pragma unroll
    for (int m = 0; m < 8; m++) v1[m] = (bf16_t)trb[tensor][ww][8 + m][nc];
    *(v8bf*)dst       = v0;
    *(v8bf*)(dst + 8) = v1;
  }
}

// ---- Kernel 2: per-channel triangle GEMM  inter_c = A_c x B_c^T -----------
// Double-buffered async global->LDS staging of 128x32 A/B panels.
// 8 waves, each 32x64 (2x4 WMMA subtiles). LDS rows padded to 40 halfs (80B).
#define SPAD 40
__global__ void k_triangle(const bf16_t* __restrict__ aws, const bf16_t* __restrict__ bws,
                           float* __restrict__ interws) {
  __shared__ bf16_t sA[2][128][SPAD];
  __shared__ bf16_t sB[2][128][SPAD];

  const int c    = blockIdx.x >> 4;
  const int tile = blockIdx.x & 15;
  const int i0   = (tile >> 2) * 128, j0 = (tile & 3) * 128;
  const int t = threadIdx.x, w = t >> 5, lane = t & 31, half = lane >> 4, nn = lane & 15;
  const int wi = w >> 1, wj = w & 1;
  const long cbase = (long)c * NROW;
  const bf16_t* Ag = aws + cbase;
  const bf16_t* Bg = bws + cbase;

  const int srow0 = t >> 2;   // 0..63 (second half via +64)
  const int sseg  = t & 3;    // 16B segment within a 64B k-row

  // stage: issue 4 async b128 copies per thread (=> 4 instructions per wave)
  auto stage = [&](int buf, int kb) {
    const int kbase = kb * 32 + sseg * 8;
#pragma unroll
    for (int u = 0; u < 2; u++) {
      int row = srow0 + u * 64;
      async_copy_b128((uint32_t)(uintptr_t)&sA[buf][row][sseg * 8],
                      Ag + (long)(i0 + row) * NN + kbase);
      async_copy_b128((uint32_t)(uintptr_t)&sB[buf][row][sseg * 8],
                      Bg + (long)(j0 + row) * NN + kbase);
    }
  };

  v8f acc[2][4] = {};
  stage(0, 0);
  for (int kb = 0; kb < 16; kb++) {
    const int cur = kb & 1;
    if (kb + 1 < 16) {
      stage(cur ^ 1, kb + 1);
      asm volatile("s_wait_asynccnt 0x4" ::: "memory");  // buffer `cur` complete
    } else {
      asm volatile("s_wait_asynccnt 0x0" ::: "memory");
    }
    __syncthreads();   // all waves' panel pieces visible

    v16bf af[2], bfv[4];
#pragma unroll
    for (int mi = 0; mi < 2; mi++)
      af[mi] = fragA_contig(&sA[cur][wi * 32 + mi * 16 + nn][0], half);
#pragma unroll
    for (int nj = 0; nj < 4; nj++)
      bfv[nj] = fragB_contig(&sB[cur][wj * 64 + nj * 16 + nn][0], half);
#pragma unroll
    for (int mi = 0; mi < 2; mi++)
#pragma unroll
      for (int nj = 0; nj < 4; nj++)
        acc[mi][nj] = __builtin_amdgcn_wmma_f32_16x16x32_bf16(
            false, af[mi], false, bfv[nj], (short)0, acc[mi][nj], false, false);

    __syncthreads();   // done reading buffer `cur`; next iter may overwrite it
  }

  float* dst = interws + cbase;
#pragma unroll
  for (int mi = 0; mi < 2; mi++)
#pragma unroll
    for (int nj = 0; nj < 4; nj++)
#pragma unroll
      for (int rr = 0; rr < 8; rr++) {
        int row = i0 + wi * 32 + mi * 16 + rr + 8 * half;
        int col = j0 + wj * 64 + nj * 16 + nn;
        dst[(long)row * NN + col] = acc[mi][nj][rr];
      }
}

// ---- Kernel 3: fused channel-LN + Wout projection (WMMA) + gate -----------
__global__ void k_out(const float* __restrict__ interws,
                      const float* __restrict__ niw, const float* __restrict__ nib,
                      const bf16_t* __restrict__ WoB, const float* __restrict__ gws,
                      float* __restrict__ out) {
  __shared__ float tile[16][CC];
  __shared__ float mu_s[16], rs_s[16];
  __shared__ float niw_s[CC], nib_s[CC];

  const int  t     = threadIdx.x;
  const long flat0 = (long)blockIdx.x * 16;
  if (t < CC) { niw_s[t] = niw[t]; nib_s[t] = nib[t]; }
  {
    const int f = t & 15, cb = t >> 4;
#pragma unroll
    for (int it = 0; it < 8; it++) {
      int c = cb * 8 + it;
      tile[f][c] = interws[(long)c * NROW + flat0 + f];
    }
  }
  __syncthreads();
  if (t < 16) {
    float s = 0.f, sq = 0.f;
    for (int c = 0; c < CC; c++) { float v = tile[t][c]; s += v; sq += v * v; }
    float mu  = s * (1.0f / CC);
    float var = sq * (1.0f / CC) - mu * mu;
    mu_s[t] = mu;
    rs_s[t] = rsqrtf(var + LN_EPS);
  }
  __syncthreads();

  const int w = t >> 5, lane = t & 31, half = lane >> 4, nn = lane & 15;
  const float mu = mu_s[nn], rs = rs_s[nn];
  v8f acc = {};
#pragma unroll
  for (int kb = 0; kb < 4; kb++) {
    const int kbase = kb * 32;
    v16bf af;
#pragma unroll
    for (int e = 0; e < 16; e++) {
      int K = 16 * half + e;   // A-operand K pattern
      int c = kbase + K;
      af[e] = (bf16_t)((tile[nn][c] - mu) * rs * niw_s[c] + nib_s[c]);
    }
    v16bf bf = fragW(WoB, kb, w, lane);
    acc = __builtin_amdgcn_wmma_f32_16x16x32_bf16(false, af, false, bf, (short)0, acc, false, false);
  }
#pragma unroll
  for (int rr = 0; rr < 8; rr++) {
    long row = flat0 + rr + 8 * half;
    int  col = w * 16 + nn;
    long idx = row * CC + col;
    out[idx] = acc[rr] * gws[idx];
  }
}

// ---- launch ---------------------------------------------------------------
extern "C" void kernel_launch(void* const* d_in, const int* in_sizes, int n_in,
                              void* d_out, int out_size, void* d_ws, size_t ws_size,
                              hipStream_t stream) {
  const float* pair   = (const float*)d_in[0];
  const float* norm_w = (const float*)d_in[1];
  const float* norm_b = (const float*)d_in[2];
  const float* Wa     = (const float*)d_in[3];
  const float* Wb     = (const float*)d_in[4];
  const float* Wg     = (const float*)d_in[5];
  const float* ni_w   = (const float*)d_in[6];
  const float* ni_b   = (const float*)d_in[7];
  const float* Wout   = (const float*)d_in[8];
  float*       out    = (float*)d_out;

  char* ws = (char*)d_ws;
  bf16_t* wa_bf   = (bf16_t*)(ws + 0);                 //  32 KB (fragment-major)
  bf16_t* wb_bf   = (bf16_t*)(ws + 32768);             //  32 KB
  bf16_t* wg_bf   = (bf16_t*)(ws + 65536);             //  32 KB
  bf16_t* wout_bf = (bf16_t*)(ws + 98304);             //  32 KB
  bf16_t* a_ws    = (bf16_t*)(ws + 131072);            //  64 MB  [c][i*512+k] bf16
  bf16_t* b_ws    = (bf16_t*)(ws + 131072 + 67108864); //  64 MB  [c][j*512+k] bf16
  float*  g_ws    = (float*)(ws + 131072 + 2UL * 67108864);                 // 128 MB f32
  float*  int_ws  = (float*)(ws + 131072 + 2UL * 67108864 + 134217728UL);   // 128 MB f32

  k_pack_weights<<<256, 256, 0, stream>>>(Wa, Wb, Wg, Wout, wa_bf, wb_bf, wg_bf, wout_bf);
  k_ln_proj<<<NROW / 16, 256, 0, stream>>>(pair, norm_w, norm_b, wa_bf, wb_bf, wg_bf,
                                           a_ws, b_ws, g_ws);
  k_triangle<<<CC * 16, 256, 0, stream>>>(a_ws, b_ws, int_ws);
  k_out<<<NROW / 16, 256, 0, stream>>>(int_ws, ni_w, ni_b, wout_bf, g_ws, out);
}
